// MGC_Net_43593918055150
// MI455X (gfx1250) — compile-verified, matrix operand order checked by reading, not run on hardware
//
#include <hip/hip_runtime.h>

// ---------------------------------------------------------------------------
// GAT layer for MI455X (gfx1250, wave32, WMMA)
//   N=2048, F_IN=768, F_OUT=64, H=8, ALPHA=0.2
// Pipeline:
//   K1: Wh = h @ W via V_WMMA_F32_16X16X4_F32 (f32 precision for logits),
//       also emits Wh^T in f16 for the attention GEMM's B-fragments.
//   K2: e_i/e_j per (node, head) (tiny f32 dots), stored head-major.
//   K3: fused masked-softmax + attn@Wh (flash-attention style, online
//       softmax in registers) via V_WMMA_F32_16X16X32_F16, then ELU.
// ---------------------------------------------------------------------------

#define GAT_N     2048
#define GAT_FIN   768
#define GAT_FOUT  64
#define GAT_H     8
#define GAT_NHD   (GAT_FOUT * GAT_H)   // 512
#define GAT_ALPHA 0.2f
#define GAT_NEG   -9.0e15f

typedef float     v2f  __attribute__((ext_vector_type(2)));
typedef float     v4f  __attribute__((ext_vector_type(4)));
typedef float     v8f  __attribute__((ext_vector_type(8)));
typedef _Float16  v8h  __attribute__((ext_vector_type(8)));
typedef _Float16  v16h __attribute__((ext_vector_type(16)));

// ---------------------------------------------------------------------------
// Kernel 1: Wh = h(2048x768) @ W(768x512), one 16x16 tile per wave.
// f32 WMMA, K=4 per instruction, 192 MACs of depth per tile.
// Outputs: Wh   (f32, row-major [2048][512])  -> used by kernel 2
//          WhT  (f16, [512][2048], d-major)   -> B-fragments in kernel 3
// ---------------------------------------------------------------------------
__global__ __launch_bounds__(256) void gat_gemm1_kernel(
    const float* __restrict__ hmat, const float* __restrict__ W,
    float* __restrict__ Wh, _Float16* __restrict__ WhT)
{
    const int lane = threadIdx.x & 31;
    const int wid  = blockIdx.x * 8 + (threadIdx.x >> 5);  // 0..4095
    const int tm   = wid >> 5;   // 0..127 (row tile)
    const int tn   = wid & 31;   // 0..31  (col tile)
    const int m    = lane & 15;
    const int hi   = lane >> 4;  // lane group 0/1

    // A frag (16x4 f32): lane holds row m; VGPR r -> K = k + 2*hi + r
    const float* ap = hmat + (size_t)(tm * 16 + m) * GAT_FIN + 2 * hi;
    // B frag (4x16 f32): lane holds col m; VGPR r -> K = k + 2*hi + r
    const float* bp = W + (size_t)(2 * hi) * GAT_NHD + tn * 16 + m;

    v8f c = {};
#pragma unroll 4
    for (int k = 0; k < GAT_FIN; k += 4) {
        v2f a = *(const v2f*)(ap + k);
        v2f b;
        b[0] = bp[(size_t)k * GAT_NHD];
        b[1] = bp[(size_t)(k + 1) * GAT_NHD];
        c = __builtin_amdgcn_wmma_f32_16x16x4_f32(
                false, a, false, b, (short)0, c, false, false);
    }

    // C layout: VGPR r, lane L -> (row r + 8*hi, col L%16)
    const int rowbase = tm * 16 + 8 * hi;
    const int coln    = tn * 16 + m;
    v8h hv;
#pragma unroll
    for (int r = 0; r < 8; ++r) {
        Wh[(size_t)(rowbase + r) * GAT_NHD + coln] = c[r];
        hv[r] = (_Float16)c[r];
    }
    // WhT[col][row]: 8 consecutive rows -> one b128 store per lane
    *(v8h*)(WhT + (size_t)coln * GAT_N + rowbase) = hv;
}

// ---------------------------------------------------------------------------
// Kernel 2: e_i[n,h] = <Wh[n,h,:], a_i[h]>, e_j likewise. Head-major outputs.
// ---------------------------------------------------------------------------
__global__ __launch_bounds__(256) void gat_e_kernel(
    const float* __restrict__ Wh, const float* __restrict__ a,
    float* __restrict__ ei_t, float* __restrict__ ej_t)
{
    const int idx = blockIdx.x * 256 + threadIdx.x;  // 0..16383
    const int n = idx >> 3;
    const int h = idx & 7;
    const float* wp = Wh + (size_t)n * GAT_NHD + h * GAT_FOUT;
    const float* ai = a + h * (2 * GAT_FOUT);
    const float* aj = ai + GAT_FOUT;
    float si = 0.0f, sj = 0.0f;
#pragma unroll 8
    for (int d = 0; d < GAT_FOUT; ++d) {
        float w = wp[d];
        si += w * ai[d];
        sj += w * aj[d];
    }
    ei_t[h * GAT_N + n] = si;
    ej_t[h * GAT_N + n] = sj;
}

// ---------------------------------------------------------------------------
// Kernel 3: fused masked softmax + aggregation + ELU.
// One wave owns a (16-row i-block, head): 4 x v8f accumulators (16x64 out),
// online softmax state (m,l) per row held per-lane, P tiles (16x32 f16) built
// directly in A-fragment layout, B from WhT with contiguous b128 loads.
// Block = 256 threads = 8 waves = all 8 heads of one i-block.
// ---------------------------------------------------------------------------
__global__ __launch_bounds__(256) void gat_attn_kernel(
    const float* __restrict__ adj, const _Float16* __restrict__ WhT,
    const float* __restrict__ ei_t, const float* __restrict__ ej_t,
    float* __restrict__ out)
{
    const int lane = threadIdx.x & 31;
    const int head = threadIdx.x >> 5;   // 0..7
    const int iblk = blockIdx.x;         // 0..127
    const int m    = lane & 15;
    const int hi   = lane >> 4;
    const int row_i = iblk * 16 + m;

    const float     ei     = ei_t[head * GAT_N + row_i];
    const float*    adjrow = adj + (size_t)row_i * GAT_N;
    const float*    ejrow  = ej_t + (size_t)head * GAT_N;
    const _Float16* whbase = WhT + (size_t)(head * GAT_FOUT) * GAT_N;

    float mrun = -INFINITY;   // running row max (row = lane%16)
    float lrun = 0.0f;        // running row denominator
    v8f acc0 = {}, acc1 = {}, acc2 = {}, acc3 = {};

    for (int j0 = 0; j0 < GAT_N; j0 += 32) {
        // A-frag j-locals for half t: j = j0 + t + (t>=8 ? 8:0) + 8*hi
        const int o1 = j0 + 8 * hi;        // halves 0..7
        const int o2 = j0 + 16 + 8 * hi;   // halves 8..15

        v4f ad0 = *(const v4f*)(adjrow + o1);
        v4f ad1 = *(const v4f*)(adjrow + o1 + 4);
        v4f ad2 = *(const v4f*)(adjrow + o2);
        v4f ad3 = *(const v4f*)(adjrow + o2 + 4);
        v4f ejv0 = *(const v4f*)(ejrow + o1);
        v4f ejv1 = *(const v4f*)(ejrow + o1 + 4);
        v4f ejv2 = *(const v4f*)(ejrow + o2);
        v4f ejv3 = *(const v4f*)(ejrow + o2 + 4);

        float s[16];
#pragma unroll
        for (int t = 0; t < 4; ++t) {
            float x0 = ei + ejv0[t]; x0 = x0 > 0.0f ? x0 : GAT_ALPHA * x0;
            float x1 = ei + ejv1[t]; x1 = x1 > 0.0f ? x1 : GAT_ALPHA * x1;
            float x2 = ei + ejv2[t]; x2 = x2 > 0.0f ? x2 : GAT_ALPHA * x2;
            float x3 = ei + ejv3[t]; x3 = x3 > 0.0f ? x3 : GAT_ALPHA * x3;
            s[t]      = ad0[t] > 0.0f ? x0 : GAT_NEG;
            s[t + 4]  = ad1[t] > 0.0f ? x1 : GAT_NEG;
            s[t + 8]  = ad2[t] > 0.0f ? x2 : GAT_NEG;
            s[t + 12] = ad3[t] > 0.0f ? x3 : GAT_NEG;
        }

        // online softmax: row spread over lane L and L^16
        float cmax = s[0];
#pragma unroll
        for (int t = 1; t < 16; ++t) cmax = fmaxf(cmax, s[t]);
        cmax = fmaxf(cmax, __shfl_xor(cmax, 16, 32));
        const float mnew  = fmaxf(mrun, cmax);        // always finite (>= GAT_NEG)
        const float scale = __expf(mrun - mnew);      // 0 on first real chunk

        float p[16];
        float csum = 0.0f;
#pragma unroll
        for (int t = 0; t < 16; ++t) {
            p[t] = __expf(s[t] - mnew);
            csum += p[t];
        }
        csum += __shfl_xor(csum, 16, 32);
        lrun = lrun * scale + csum;
        mrun = mnew;

        // rescale accumulators: acc row = r + 8*hi, scale lives at lane (row%16)
#pragma unroll
        for (int r = 0; r < 8; ++r) {
            float sr = __shfl(scale, r + 8 * hi, 32);
            acc0[r] *= sr; acc1[r] *= sr; acc2[r] *= sr; acc3[r] *= sr;
        }

        // P tile -> A fragment (already in layout order)
        v16h af;
#pragma unroll
        for (int t = 0; t < 16; ++t) af[t] = (_Float16)p[t];

        // 4 d-tiles of B (32x16 f16) from WhT: contiguous along K
#pragma unroll
        for (int dt = 0; dt < 4; ++dt) {
            const _Float16* bp = whbase + (size_t)(dt * 16 + m) * GAT_N;
            v8h blo = *(const v8h*)(bp + o1);
            v8h bhi = *(const v8h*)(bp + o2);
            v16h bf;
#pragma unroll
            for (int t = 0; t < 8; ++t) { bf[t] = blo[t]; bf[t + 8] = bhi[t]; }
            if (dt == 0)
                acc0 = __builtin_amdgcn_wmma_f32_16x16x32_f16(
                           false, af, false, bf, (short)0, acc0, false, false);
            else if (dt == 1)
                acc1 = __builtin_amdgcn_wmma_f32_16x16x32_f16(
                           false, af, false, bf, (short)0, acc1, false, false);
            else if (dt == 2)
                acc2 = __builtin_amdgcn_wmma_f32_16x16x32_f16(
                           false, af, false, bf, (short)0, acc2, false, false);
            else
                acc3 = __builtin_amdgcn_wmma_f32_16x16x32_f16(
                           false, af, false, bf, (short)0, acc3, false, false);
        }
    }

    // epilogue: divide by row denominator, ELU, store
#pragma unroll
    for (int r = 0; r < 8; ++r) {
        const float lr  = __shfl(lrun, r + 8 * hi, 32);
        const float inv = 1.0f / lr;
        const int   orow = iblk * 16 + r + 8 * hi;
        float* orp = out + (size_t)orow * GAT_NHD + head * GAT_FOUT + m;
        float v0 = acc0[r] * inv; v0 = v0 > 0.0f ? v0 : (__expf(v0) - 1.0f);
        float v1 = acc1[r] * inv; v1 = v1 > 0.0f ? v1 : (__expf(v1) - 1.0f);
        float v2 = acc2[r] * inv; v2 = v2 > 0.0f ? v2 : (__expf(v2) - 1.0f);
        float v3 = acc3[r] * inv; v3 = v3 > 0.0f ? v3 : (__expf(v3) - 1.0f);
        orp[0]  = v0;
        orp[16] = v1;
        orp[32] = v2;
        orp[48] = v3;
    }
}

// ---------------------------------------------------------------------------
// Launch
// ---------------------------------------------------------------------------
extern "C" void kernel_launch(void* const* d_in, const int* in_sizes, int n_in,
                              void* d_out, int out_size, void* d_ws, size_t ws_size,
                              hipStream_t stream)
{
    (void)in_sizes; (void)n_in; (void)out_size; (void)ws_size;
    const float* hmat = (const float*)d_in[0];   // (2048, 768)
    const float* adj  = (const float*)d_in[1];   // (2048, 2048)
    const float* W    = (const float*)d_in[2];   // (768, 512)
    const float* a    = (const float*)d_in[3];   // (1, 8, 128)
    float* out = (float*)d_out;                  // (2048, 512)

    // workspace: Wh f32 (4MB) | WhT f16 (2MB) | ei_t (64KB) | ej_t (64KB)
    float*    Wh   = (float*)d_ws;
    _Float16* WhT  = (_Float16*)((char*)d_ws + (size_t)GAT_N * GAT_NHD * 4);
    float*    ei_t = (float*)((char*)WhT + (size_t)GAT_NHD * GAT_N * 2);
    float*    ej_t = ei_t + GAT_H * GAT_N;

    // K1: 4096 tiles, 8 waves/block -> 512 blocks
    gat_gemm1_kernel<<<512, 256, 0, stream>>>(hmat, W, Wh, WhT);
    // K2: 16384 (node, head) pairs
    gat_e_kernel<<<64, 256, 0, stream>>>(Wh, a, ei_t, ej_t);
    // K3: 128 i-blocks x 8 heads (one wave per head)
    gat_attn_kernel<<<128, 256, 0, stream>>>(adj, WhT, ei_t, ej_t, out);
}
